// QMogrifierStack_6511170421080
// MI455X (gfx1250) — compile-verified, matrix operand order checked by reading, not run on hardware
//
#include <hip/hip_runtime.h>
#include <hip/hip_bf16.h>

// MI455X / gfx1250: wave32, WMMA bf16 16x16x32, fp32 accumulate.
typedef __attribute__((ext_vector_type(16))) __bf16 v16bf;
typedef __attribute__((ext_vector_type(8)))  __bf16 bf16x8;
typedef __attribute__((ext_vector_type(8)))  float  v8f;

#define BB 64
#define SS 512
#define HH 1024
#define NBLK 64   // persistent grid: 64 blocks x 128 threads (4 waves) — trivially co-resident

// Build a v16bf fragment from two contiguous 16B chunks (K-packed layout):
// lane group g = lane>>4 reads K = kBase + g*8 .. +7 (elems 0..7) and +16 (elems 8..15),
// matching the CDNA5 16-bit A/B VGPR layout (V0-3 / V4-7 K sub-blocks).
__device__ __forceinline__ v16bf load_frag16(const __bf16* __restrict__ p) {
    bf16x8 lo = *reinterpret_cast<const bf16x8*>(p);
    bf16x8 hi = *reinterpret_cast<const bf16x8*>(p + 16);
    return __builtin_shufflevector(lo, hi, 0,1,2,3,4,5,6,7,8,9,10,11,12,13,14,15);
}

__device__ __forceinline__ float sigm_(float x) { return 1.0f / (1.0f + __expf(-x)); }
__device__ __forceinline__ float tanh_(float x) {
    float e2 = __expf(-2.0f * fabsf(x));
    float t  = (1.0f - e2) / (1.0f + e2);
    return copysignf(t, x);
}
__device__ __forceinline__ v8f wmma_bf16(v16bf a, v16bf b, v8f c) {
    return __builtin_amdgcn_wmma_f32_16x16x32_bf16(false, a, false, b, (short)0, c, false, false);
}

// Device-scope grid barrier: monotonically increasing arrive counter (zeroed per layer
// by the init kernel), release/acquire fences at agent scope, s_sleep in the spin.
__device__ __forceinline__ void grid_barrier(unsigned* bar, unsigned target) {
    __syncthreads();
    if (threadIdx.x == 0) {
        __builtin_amdgcn_fence(__ATOMIC_RELEASE, "agent");
        __hip_atomic_fetch_add(bar, 1u, __ATOMIC_RELAXED, __HIP_MEMORY_SCOPE_AGENT);
        while (__hip_atomic_load(bar, __ATOMIC_RELAXED, __HIP_MEMORY_SCOPE_AGENT) < target)
            __builtin_amdgcn_s_sleep(1);
        __builtin_amdgcn_fence(__ATOMIC_ACQUIRE, "agent");
    }
    __syncthreads();
}

// ---------------- weight packing (fp32 -> bf16, once per launch) ----------------

// Wcat[row, 0:inl) = Wih[row,:], Wcat[row, inl:inl+H) = Whh[row,:]; row-major, stride K.
__global__ void k_pack_wcat(const float* __restrict__ Wih, const float* __restrict__ Whh,
                            __bf16* __restrict__ Wcat, int inl, int K) {
    const int total = 4 * HH * K;
    for (int idx = blockIdx.x * blockDim.x + threadIdx.x; idx < total;
         idx += gridDim.x * blockDim.x) {
        int row = idx / K, k = idx - row * K;
        float v = (k < inl) ? Wih[(size_t)row * inl + k] : Whh[(size_t)row * HH + (k - inl)];
        Wcat[idx] = (__bf16)v;
    }
}

// modWb[j, 0:1024) = modW[j, 0:1024) (the h-part of the mogrifier matrix), stride 1024.
__global__ void k_pack_modw(const float* __restrict__ modW, __bf16* __restrict__ modWb, int inl) {
    const int total = inl * HH;
    for (int idx = blockIdx.x * blockDim.x + threadIdx.x; idx < total;
         idx += gridDim.x * blockDim.x) {
        int row = idx >> 10, k = idx & 1023;
        modWb[idx] = (__bf16)modW[(size_t)row * (HH + 2) + k];
    }
}

// zero h-region of A-panel, cell state c, and the grid-barrier counter at layer start
__global__ void k_init_layer(__bf16* __restrict__ Ah, int Kst, float* __restrict__ c,
                             unsigned* __restrict__ bar) {
    if (blockIdx.x == 0 && threadIdx.x == 0) *bar = 0u;
    const int total = BB * HH;
    for (int idx = blockIdx.x * blockDim.x + threadIdx.x; idx < total;
         idx += gridDim.x * blockDim.x) {
        int b = idx >> 10, j = idx & 1023;
        Ah[(size_t)b * Kst + j] = (__bf16)0.0f;
        c[idx] = 0.0f;
    }
}

// ---------------- persistent per-layer kernel ----------------
// One launch per layer; internal loop over S timesteps with a device-wide barrier
// between the two dependent phases. Eliminates ~2000 serial kernel-launch latencies.
//
// Phase A (blocks 0..inl/64-1): m = sigmoid(h @ modWb^T + dt*w_d + lq*w_lq + modb);
//                               A-panel x-region <- bf16(m * x_t).
// Phase B (all 64 blocks):      gates = A(64 x Kst) @ Wcat^T with the wave owning
//                               N-tiles {j, H+j, 2H+j, 3H+j} so i/f/g/o sit in matching
//                               register slots; full LSTM cell update in-register.
__global__ __launch_bounds__(128)
void k_layer_persistent(__bf16* __restrict__ Apanel,        // 64 x Kst ([m*x | h])
                        const __bf16* __restrict__ modWb,   // inl x 1024 bf16
                        const float*  __restrict__ modW,    // inl x 1026 fp32 (dt/lq cols)
                        const float*  __restrict__ modb,    // inl
                        const __bf16* __restrict__ Wcat,    // 4096 x Kst bf16
                        const float*  __restrict__ bih, const float* __restrict__ bhh,
                        const float*  __restrict__ dseq, const float* __restrict__ lqseq,
                        const float*  __restrict__ xsrc, int xst,
                        float*  __restrict__ c,             // 64 x 1024
                        float*  __restrict__ hout,          // seq dest fp32, stride H
                        float*  __restrict__ lasth,         // null for layer 0
                        int inl, unsigned* __restrict__ bar) {
    const int Kst  = inl + HH;
    const int lane = threadIdx.x & 31;
    const int wv   = threadIdx.x >> 5;
    const int grp  = lane >> 4, l16 = lane & 15;
    const int nModBlk = inl >> 6;

    // ---- phase-A (mod) fixed addressing: wave handles 1 N-tile x 4 M-tiles
    const int     ntile  = blockIdx.x * 4 + wv;
    const __bf16* m_arow = Apanel + inl + (size_t)l16 * Kst + grp * 8;
    const __bf16* m_bcol = modWb + (size_t)(ntile * 16 + l16) * HH + grp * 8;

    // ---- phase-B (gates) fixed addressing
    const int     j0     = blockIdx.x * 16;
    const __bf16* g_arow = Apanel + (size_t)(wv * 16 + l16) * Kst + grp * 8;
    const size_t  ko     = (size_t)grp * 8;
    const __bf16* wbi = Wcat + (size_t)(0 * HH + j0 + l16) * Kst + ko;
    const __bf16* wbf = Wcat + (size_t)(1 * HH + j0 + l16) * Kst + ko;
    const __bf16* wbg = Wcat + (size_t)(2 * HH + j0 + l16) * Kst + ko;
    const __bf16* wbo = Wcat + (size_t)(3 * HH + j0 + l16) * Kst + ko;
    const int   j   = j0 + l16;
    const float b_i = bih[j]          + bhh[j];
    const float b_f = bih[HH + j]     + bhh[HH + j];
    const float b_g = bih[2 * HH + j] + bhh[2 * HH + j];
    const float b_o = bih[3 * HH + j] + bhh[3 * HH + j];

    unsigned phase = 0;
    for (int t = 0; t < SS; ++t) {
        // ================= phase A: mogrifier gate + pack m*x =================
        if (blockIdx.x < nModBlk) {
            v8f acc[4] = {};
#pragma unroll 4
            for (int kb = 0; kb < HH; kb += 32) {
                v16bf bfr = load_frag16(m_bcol + kb);
                __builtin_prefetch(m_bcol + kb + 256, 0, 2);
#pragma unroll
                for (int mt = 0; mt < 4; ++mt) {
                    v16bf a = load_frag16(m_arow + (size_t)mt * 16 * Kst + kb);
                    acc[mt] = wmma_bf16(a, bfr, acc[mt]);
                }
            }
            const int   jm   = ntile * 16 + l16;
            const float cdt  = modW[(size_t)jm * (HH + 2) + HH];
            const float clq  = modW[(size_t)jm * (HH + 2) + HH + 1];
            const float bias = modb[jm];
#pragma unroll
            for (int mt = 0; mt < 4; ++mt) {
#pragma unroll
                for (int v = 0; v < 8; ++v) {
                    int   b  = mt * 16 + v + (grp << 3);  // C/D layout: M = v + 8*(lane/16)
                    float dt = dseq[b * SS + t];
                    float lq = lqseq[b * SS + t];
                    float m  = sigm_(acc[mt][v] + dt * cdt + lq * clq + bias);
                    float x  = xsrc[((size_t)b * SS + t) * xst + jm];
                    Apanel[(size_t)b * Kst + jm] = (__bf16)(m * x);
                }
            }
        }
        grid_barrier(bar, (++phase) * NBLK);

        // ================= phase B: gates GEMM + fused LSTM cell =================
        {
            v8f ai = {}, af = {}, ag = {}, ao = {};
#pragma unroll 2
            for (int kb = 0; kb < Kst; kb += 32) {
                v16bf a  = load_frag16(g_arow + kb);       // A reused across 4 WMMAs
                v16bf fi = load_frag16(wbi + kb);
                v16bf ff = load_frag16(wbf + kb);
                v16bf fg = load_frag16(wbg + kb);
                v16bf fo = load_frag16(wbo + kb);
                __builtin_prefetch(wbi + kb + 256, 0, 2);
                __builtin_prefetch(wbg + kb + 256, 0, 2);
                ai = wmma_bf16(a, fi, ai);
                af = wmma_bf16(a, ff, af);
                ag = wmma_bf16(a, fg, ag);
                ao = wmma_bf16(a, fo, ao);
            }
#pragma unroll
            for (int v = 0; v < 8; ++v) {
                int   b    = wv * 16 + v + (grp << 3);
                float ig   = sigm_(ai[v] + b_i);
                float fgv  = sigm_(af[v] + b_f);
                float gg   = tanh_(ag[v] + b_g);
                float og   = sigm_(ao[v] + b_o);
                float cold = c[(size_t)b * HH + j];
                float cn   = fgv * cold + ig * gg;
                float hn   = og * tanh_(cn);
                c[(size_t)b * HH + j] = cn;
                hout[((size_t)b * SS + t) * HH + j] = hn;
                Apanel[(size_t)b * Kst + inl + j] = (__bf16)hn;   // h for next step
                if (lasth && t == SS - 1) lasth[(size_t)b * HH + j] = hn;
            }
        }
        grid_barrier(bar, (++phase) * NBLK);
    }
}

// ---------------- host orchestration ----------------

extern "C" void kernel_launch(void* const* d_in, const int* in_sizes, int n_in,
                              void* d_out, int out_size, void* d_ws, size_t ws_size,
                              hipStream_t stream) {
    (void)in_sizes; (void)n_in; (void)out_size; (void)ws_size;
    const float* seq   = (const float*)d_in[0];
    const float* dseq  = (const float*)d_in[1];
    const float* lqseq = (const float*)d_in[2];

    const float* modW[2], *modb[2], *Wih[2], *Whh[2], *bih[2], *bhh[2];
    const int inl[2] = {512, 1024};
    for (int l = 0; l < 2; ++l) {
        int base = 3 + 6 * l;
        modW[l] = (const float*)d_in[base + 0];
        modb[l] = (const float*)d_in[base + 1];
        Wih[l]  = (const float*)d_in[base + 2];
        Whh[l]  = (const float*)d_in[base + 3];
        bih[l]  = (const float*)d_in[base + 4];
        bhh[l]  = (const float*)d_in[base + 5];
    }

    // workspace carve (256B aligned); total ~33 MB — bf16 weights stay L2-resident (192MB).
    char* p = (char*)d_ws;
    auto carve = [&](size_t bytes) { char* r = p; p += (bytes + 255) & ~(size_t)255; return r; };
    __bf16* Wcat[2];
    Wcat[0] = (__bf16*)carve((size_t)4096 * 1536 * 2);
    Wcat[1] = (__bf16*)carve((size_t)4096 * 2048 * 2);
    __bf16* mwb[2];
    mwb[0] = (__bf16*)carve((size_t)512 * 1024 * 2);
    mwb[1] = (__bf16*)carve((size_t)1024 * 1024 * 2);
    __bf16*   Apanel = (__bf16*)carve((size_t)BB * 2048 * 2);
    float*    cstate = (float*)carve((size_t)BB * HH * 4);
    unsigned* bar    = (unsigned*)carve(256);

    float* seqout = (float*)d_out;                       // (B,S,H) fp32
    float* lasth  = seqout + (size_t)BB * SS * HH;       // (B,H)   fp32

    // one-time (per call) fp32 -> bf16 weight packing
    k_pack_wcat<<<1024, 256, 0, stream>>>(Wih[0], Whh[0], Wcat[0], 512, 1536);
    k_pack_wcat<<<1024, 256, 0, stream>>>(Wih[1], Whh[1], Wcat[1], 1024, 2048);
    k_pack_modw<<<256, 256, 0, stream>>>(modW[0], mwb[0], 512);
    k_pack_modw<<<256, 256, 0, stream>>>(modW[1], mwb[1], 1024);

    for (int l = 0; l < 2; ++l) {
        const int Kst = inl[l] + HH;
        // layer 0 reads d_in seq; layer 1 reads layer-0 output already in d_out
        // (phase A reads (b,t) before phase B of the same step overwrites it).
        const float* xsrc = l ? seqout : seq;
        const int    xst  = l ? HH : 512;

        k_init_layer<<<256, 256, 0, stream>>>(Apanel + inl[l], Kst, cstate, bar);
        k_layer_persistent<<<NBLK, 128, 0, stream>>>(
            Apanel, mwb[l], modW[l], modb[l], Wcat[l], bih[l], bhh[l],
            dseq, lqseq, xsrc, xst, cstate, seqout,
            (l == 1) ? lasth : nullptr, inl[l], bar);
    }
}
